// GroupEncoder_86835648791131
// MI455X (gfx1250) — matrix operand order repaired and checked by reading, not compiled, not run on hardware
//
#include <hip/hip_runtime.h>
#include <math.h>

#define BREFL 1000000
#define DIN   64
#define HID   128
#define KGRP  4096
#define ALPHA_MIN 0.1f

// Use inline-asm GLOBAL_LOAD_ASYNC_TO_LDS_B128 for the x-tile staging.
// Flip to 0 to fall back to synchronous float4 loads + ds_store.
#define USE_ASYNC_LDS 1

typedef __attribute__((ext_vector_type(16))) __bf16 v16bf;
typedef __attribute__((ext_vector_type(8)))  float  v8f;

union FragU { uint4 q[2]; v16bf v; };

__device__ __forceinline__ __bf16 f2bf(float f) {
  unsigned u = __builtin_bit_cast(unsigned, f);
  unsigned r = u + 0x7FFFu + ((u >> 16) & 1u);   // round-to-nearest-even
  unsigned short s = (unsigned short)(r >> 16);
  return __builtin_bit_cast(__bf16, s);
}

__device__ __forceinline__ float siluf(float x) {
  return x / (1.0f + __expf(-x));
}

// A-fragment (16x32 bf16) built from an fp32 LDS tile:
// lanes 0-15 row=lane hold K=k0+[0..7] (V0-3) and k0+[16..23] (V4-7);
// lanes 16-31 row=lane-16 hold K=k0+[8..15] and k0+[24..31].
__device__ __forceinline__ v16bf load_fragA_f32(const float* base, int stride,
                                                int r0, int k0, int lane) {
  int r = r0 + (lane & 15);
  int koff = (lane & 16) ? 8 : 0;
  const float* p = base + r * stride + k0 + koff;
  v16bf a;
#pragma unroll
  for (int i = 0; i < 8; ++i) {
    a[i]     = f2bf(p[i]);
    a[8 + i] = f2bf(p[16 + i]);
  }
  return a;
}

// A-fragment directly from a bf16 tile (two 16B chunks per lane).
__device__ __forceinline__ v16bf load_fragA(const __bf16* base, int stride,
                                            int r0, int k0, int lane) {
  int r = r0 + (lane & 15);
  int koff = (lane & 16) ? 8 : 0;
  const __bf16* p = base + r * stride + k0 + koff;
  FragU f;
  f.q[0] = *(const uint4*)(p);
  f.q[1] = *(const uint4*)(p + 16);
  return f.v;
}

// B-fragment (32x16 bf16), W stored transposed [n][k]: lanes 0-15 col=lane hold
// K=k0+[0..7],[16..23]; lanes 16-31 col=lane-16 hold K=k0+[8..15],[24..31].
__device__ __forceinline__ v16bf load_fragB(const __bf16* base, int stride,
                                            int n0, int k0, int lane) {
  int n = n0 + (lane & 15);
  int kk = k0 + ((lane & 16) ? 8 : 0);
  const __bf16* p = base + n * stride + kk;
  FragU f;
  f.q[0] = *(const uint4*)(p);
  f.q[1] = *(const uint4*)(p + 16);
  return f.v;
}

// ---------------- weight prep: transpose + fp32 -> bf16 ----------------
__global__ void prep_weights(const float* __restrict__ W1,
                             const float* __restrict__ W2,
                             __bf16* __restrict__ w1t,
                             __bf16* __restrict__ w2t) {
  int idx = blockIdx.x * blockDim.x + threadIdx.x;
  if (idx < HID * DIN) {                 // w1t[n][k] = W1[k][n], n<128, k<64
    int n = idx / DIN, k = idx % DIN;
    w1t[idx] = f2bf(W1[k * HID + n]);
  } else if (idx < HID * DIN + HID * HID) {
    int j = idx - HID * DIN;             // w2t[n][k] = W2[k][n]
    int n = j / HID, k = j % HID;
    w2t[j] = f2bf(W2[k * HID + n]);
  }
}

__global__ void zero_ws(float* __restrict__ p, int n) {
  int i = blockIdx.x * blockDim.x + threadIdx.x;
  if (i < n) p[i] = 0.0f;
}

__global__ void count_kernel(const int* __restrict__ labels,
                             float* __restrict__ counts) {
  int i = blockIdx.x * blockDim.x + threadIdx.x;
  if (i < BREFL) atomicAdd(&counts[labels[i]], 1.0f);
}

// ---------------- phi: z = silu(silu(xW1+b1)W2+b2), scatter into sums ------
__global__ __launch_bounds__(256) void phi_kernel(
    const float* __restrict__ x, const int* __restrict__ labels,
    const float* __restrict__ b1, const float* __restrict__ b2,
    const __bf16* __restrict__ w1t, const __bf16* __restrict__ w2t,
    float* __restrict__ sums) {
  __shared__ float  lds_xf[128 * 68];   // 128 rows x 64 fp32 (stride 68)
  __shared__ __bf16 lds_z[128 * 136];   // 128 rows x 128 bf16 (stride 136)

  const int tid  = threadIdx.x;
  const int lane = tid & 31;
  const int wave = tid >> 5;
  const int base = blockIdx.x * 128;

  // ---- stage x tile into LDS (fp32) ----
#if USE_ASYNC_LDS
#pragma unroll
  for (int it = 0; it < 8; ++it) {
    int j   = tid + it * 256;           // 2048 float4 total
    int row = j >> 4;
    int c4  = (j & 15) * 4;
    int gr  = base + row;
    // OOB rows redirect to row 0: their outputs are never scattered.
    int sr  = (gr < BREFL) ? gr : 0;
    const float* gp = x + (size_t)sr * DIN + c4;
    unsigned lds_off = (unsigned)(size_t)
        (__attribute__((address_space(3))) const float*)&lds_xf[row * 68 + c4];
    unsigned long long ga = (unsigned long long)(size_t)gp;
    asm volatile("global_load_async_to_lds_b128 %0, %1, off"
                 :: "v"(lds_off), "v"(ga)
                 : "memory");
  }
  asm volatile("s_wait_asynccnt 0x0" ::: "memory");
#else
#pragma unroll
  for (int it = 0; it < 8; ++it) {
    int j   = tid + it * 256;
    int row = j >> 4;
    int c4  = (j & 15) * 4;
    int gr  = base + row;
    float4 v = make_float4(0.f, 0.f, 0.f, 0.f);
    if (gr < BREFL) v = *(const float4*)(x + (size_t)gr * DIN + c4);
    *(float4*)&lds_xf[row * 68 + c4] = v;
  }
#endif
  __syncthreads();

  const int m0   = wave * 16;
  const int colL = lane & 15;
  const int rsel = (lane & 16) ? 8 : 0;

  // ---- GEMM1: 16x128 strip, K=64 ----
  v16bf a0 = load_fragA_f32(lds_xf, 68, m0, 0, lane);
  v16bf a1 = load_fragA_f32(lds_xf, 68, m0, 32, lane);
#pragma unroll
  for (int nt = 0; nt < 8; ++nt) {
    v16bf bB0 = load_fragB(w1t, DIN, nt * 16, 0, lane);
    v16bf bB1 = load_fragB(w1t, DIN, nt * 16, 32, lane);
    v8f acc = {0.f, 0.f, 0.f, 0.f, 0.f, 0.f, 0.f, 0.f};
    acc = __builtin_amdgcn_wmma_f32_16x16x32_bf16(false, a0, false, bB0,
                                                  (short)0, acc, false, false);
    acc = __builtin_amdgcn_wmma_f32_16x16x32_bf16(false, a1, false, bB1,
                                                  (short)0, acc, false, false);
    int col = nt * 16 + colL;
    float bias = b1[col];
#pragma unroll
    for (int i = 0; i < 8; ++i) {
      float zv = siluf(acc[i] + bias);
      lds_z[(m0 + rsel + i) * 136 + col] = f2bf(zv);
    }
  }
  __syncthreads();

  // ---- GEMM2: 16x128 strip, K=128 ----
  v16bf c0 = load_fragA(lds_z, 136, m0, 0, lane);
  v16bf c1 = load_fragA(lds_z, 136, m0, 32, lane);
  v16bf c2 = load_fragA(lds_z, 136, m0, 64, lane);
  v16bf c3 = load_fragA(lds_z, 136, m0, 96, lane);

  int  labv[8];
  bool val[8];
#pragma unroll
  for (int i = 0; i < 8; ++i) {
    int gr  = base + m0 + rsel + i;
    val[i]  = (gr < BREFL);
    labv[i] = val[i] ? labels[gr] : 0;
  }

#pragma unroll
  for (int nt = 0; nt < 8; ++nt) {
    v16bf bB0 = load_fragB(w2t, HID, nt * 16, 0, lane);
    v16bf bB1 = load_fragB(w2t, HID, nt * 16, 32, lane);
    v16bf bB2 = load_fragB(w2t, HID, nt * 16, 64, lane);
    v16bf bB3 = load_fragB(w2t, HID, nt * 16, 96, lane);
    v8f acc = {0.f, 0.f, 0.f, 0.f, 0.f, 0.f, 0.f, 0.f};
    acc = __builtin_amdgcn_wmma_f32_16x16x32_bf16(false, c0, false, bB0,
                                                  (short)0, acc, false, false);
    acc = __builtin_amdgcn_wmma_f32_16x16x32_bf16(false, c1, false, bB1,
                                                  (short)0, acc, false, false);
    acc = __builtin_amdgcn_wmma_f32_16x16x32_bf16(false, c2, false, bB2,
                                                  (short)0, acc, false, false);
    acc = __builtin_amdgcn_wmma_f32_16x16x32_bf16(false, c3, false, bB3,
                                                  (short)0, acc, false, false);
    int col = nt * 16 + colL;
    float bias = b2[col];
#pragma unroll
    for (int i = 0; i < 8; ++i) {
      float zv = siluf(acc[i] + bias);
      if (val[i]) atomicAdd(&sums[labv[i] * HID + col], zv);
    }
  }
}

// ---------------- mean pool ----------------
__global__ void mean_kernel(float* __restrict__ sums,
                            const float* __restrict__ counts) {
  int i = blockIdx.x * blockDim.x + threadIdx.x;
  if (i < KGRP * HID) sums[i] = sums[i] / fmaxf(counts[i >> 7], 1.0f);
}

// ---------------- rho hidden: h = silu(gf @ Wr + br) ----------------
__global__ void rho_h_kernel(const float* __restrict__ gf,
                             const float* __restrict__ Wr,
                             const float* __restrict__ br,
                             float* __restrict__ hbuf) {
  int idx = blockIdx.x * blockDim.x + threadIdx.x;
  if (idx >= KGRP * HID) return;
  int g = idx >> 7, j = idx & 127;
  float acc = br[j];
  const float* row = gf + g * HID;
#pragma unroll 4
  for (int k = 0; k < HID; ++k) acc = fmaf(row[k], Wr[k * HID + j], acc);
  hbuf[idx] = siluf(acc);
}

// ---------------- heads + gamma sample + tau ----------------
__device__ __forceinline__ unsigned pcg_next(unsigned& s) {
  s = s * 747796405u + 2891336453u;
  unsigned w = ((s >> ((s >> 28) + 4u)) ^ s) * 277803737u;
  return (w >> 22) ^ w;
}
__device__ __forceinline__ float unif01(unsigned& s) {
  return ((float)(pcg_next(s) >> 8) + 0.5f) * (1.0f / 16777216.0f);
}
__device__ float gamma_sample(float alpha, unsigned seed) {
  unsigned s = seed * 0x9E3779B9u + 42u * 2654435761u + 1u;
  (void)pcg_next(s);
  float boost = 1.0f;
  if (alpha < 1.0f) {
    float u = unif01(s);
    boost = __powf(u, 1.0f / alpha);
    alpha += 1.0f;
  }
  float d = alpha - (1.0f / 3.0f);
  float c = rsqrtf(9.0f * d);
  for (int it = 0; it < 64; ++it) {
    float u1 = unif01(s), u2 = unif01(s);
    float n = sqrtf(-2.0f * __logf(u1)) * __cosf(6.28318530718f * u2);
    float v = 1.0f + c * n;
    if (v <= 0.0f) continue;
    v = v * v * v;
    float u  = unif01(s);
    float n2 = n * n;
    if (u < 1.0f - 0.0331f * n2 * n2) return boost * d * v;
    if (__logf(u) < 0.5f * n2 + d * (1.0f - v + __logf(v))) return boost * d * v;
  }
  return boost * d;
}
__device__ __forceinline__ float softplusf(float x) {
  return (x > 20.0f) ? x : log1pf(__expf(x));
}

__global__ void head_kernel(const float* __restrict__ hbuf,
                            const float* __restrict__ wa, const float* __restrict__ ba,
                            const float* __restrict__ wb, const float* __restrict__ bb,
                            float* __restrict__ out_alpha, float* __restrict__ out_beta,
                            float* __restrict__ taug) {
  int g = blockIdx.x * blockDim.x + threadIdx.x;
  if (g >= KGRP) return;
  const float* row = hbuf + g * HID;
  float da = ba[0], db = bb[0];
#pragma unroll 4
  for (int k = 0; k < HID; ++k) {
    da = fmaf(row[k], wa[k], da);
    db = fmaf(row[k], wb[k], db);
  }
  float alpha = softplusf(da) + ALPHA_MIN;
  float beta  = softplusf(db) + ALPHA_MIN;
  float gs    = gamma_sample(alpha, (unsigned)g);
  out_alpha[g] = alpha;
  out_beta[g]  = beta;
  taug[g]      = gs / beta;
}

__global__ void gather_kernel(const int* __restrict__ labels,
                              const float* __restrict__ taug,
                              float* __restrict__ out_tau) {
  int i = blockIdx.x * blockDim.x + threadIdx.x;
  if (i < BREFL) out_tau[i] = taug[labels[i]];
}

// ---------------- launcher ----------------
extern "C" void kernel_launch(void* const* d_in, const int* in_sizes, int n_in,
                              void* d_out, int out_size, void* d_ws, size_t ws_size,
                              hipStream_t stream) {
  const float* x      = (const float*)d_in[0];
  const int*   labels = (const int*)d_in[1];
  const float* W1     = (const float*)d_in[2];
  const float* b1     = (const float*)d_in[3];
  const float* W2     = (const float*)d_in[4];
  const float* b2     = (const float*)d_in[5];
  const float* Wr     = (const float*)d_in[6];
  const float* br     = (const float*)d_in[7];
  const float* wa     = (const float*)d_in[8];
  const float* ba     = (const float*)d_in[9];
  const float* wb     = (const float*)d_in[10];
  const float* bb     = (const float*)d_in[11];
  float* out = (float*)d_out;   // [alpha(K) | beta(K) | tau(B)]

  // workspace carve-up (floats)
  float*  sums   = (float*)d_ws;              // K*H
  float*  counts = sums + KGRP * HID;         // K
  float*  hbuf   = counts + KGRP;             // K*H
  float*  taug   = hbuf + KGRP * HID;         // K
  __bf16* w1t    = (__bf16*)(taug + KGRP);    // 128*64 bf16 (16B-aligned)
  __bf16* w2t    = w1t + HID * DIN;           // 128*128 bf16

  const int T = 256;
  hipLaunchKernelGGL(prep_weights, dim3((HID * DIN + HID * HID + T - 1) / T),
                     dim3(T), 0, stream, W1, W2, w1t, w2t);
  hipLaunchKernelGGL(zero_ws, dim3((KGRP * HID + KGRP + T - 1) / T), dim3(T),
                     0, stream, sums, KGRP * HID + KGRP);
  hipLaunchKernelGGL(count_kernel, dim3((BREFL + T - 1) / T), dim3(T), 0,
                     stream, labels, counts);
  hipLaunchKernelGGL(phi_kernel, dim3((BREFL + 127) / 128), dim3(T), 0, stream,
                     x, labels, b1, b2, w1t, w2t, sums);
  hipLaunchKernelGGL(mean_kernel, dim3((KGRP * HID + T - 1) / T), dim3(T), 0,
                     stream, sums, counts);
  hipLaunchKernelGGL(rho_h_kernel, dim3((KGRP * HID + T - 1) / T), dim3(T), 0,
                     stream, sums, Wr, br, hbuf);
  hipLaunchKernelGGL(head_kernel, dim3((KGRP + T - 1) / T), dim3(T), 0, stream,
                     hbuf, wa, ba, wb, bb, out, out + KGRP, taug);
  hipLaunchKernelGGL(gather_kernel, dim3((BREFL + T - 1) / T), dim3(T), 0,
                     stream, labels, taug, out + 2 * KGRP);
}
